// AxialAttentionLayer_52115133170004
// MI455X (gfx1250) — matrix operand order repaired and checked by reading, not compiled, not run on hardware
//
#include <hip/hip_runtime.h>

// MI455X (gfx1250, wave32) axial-attention kernel.
// scores = rowsum(Q) ⊗ rowsum(K) + Q·K ; A = softmax(scores) ; out = A·V
// B=32, L=8192, S=E=D=128.
//
// Strategy: memory-bound problem (Q/out = 128 MiB streams, ~11 us at 23.3 TB/s).
// Matmuls in split-precision bf16 (hi+lo) via v_wmma_f32_16x16x32_bf16:
// 3 WMMAs per fp32-equivalent product -> ~2^-16 relative error (fp32-class
// logits for the softmax) at 3/8 the cost of the native f32 16x16x4 WMMA path.
// K^T, V^T, Q staged in LDS as hi/lo bf16 (~205 KB of the WGP's 320 KB,
// requested as dynamic LDS). One 256-thread WG per (batch, 128-row Q tile);
// each wave32 owns a 16-row stripe.

typedef __attribute__((ext_vector_type(16))) __bf16 v16bf;
typedef __attribute__((ext_vector_type(8)))  __bf16 v8bf;
typedef __attribute__((ext_vector_type(8)))  float  v8f;
typedef unsigned short ushort_t;

#define B_N      32
#define L_N      8192
#define S_N      128
#define TILE_L   128
#define NTHREADS 256
#define SPAD     136                  // LDS row stride (bf16 elems): 272 B, 16B-aligned, bank-friendly

#define ARR_BYTES (S_N * SPAD * 2)    // 34816 B per bf16 tile array
#define KT_HI_OFF 0
#define KT_LO_OFF (1 * ARR_BYTES)
#define VT_HI_OFF (2 * ARR_BYTES)
#define VT_LO_OFF (3 * ARR_BYTES)
#define QP_HI_OFF (4 * ARR_BYTES)     // Q tile hi; reused for softmax P hi
#define QP_LO_OFF (5 * ARR_BYTES)     // Q tile lo; reused for softmax P lo
#define QSUM_OFF  (6 * ARR_BYTES)
#define KSUM_OFF  (6 * ARR_BYTES + 512)
#define SMEM_BYTES (6 * ARR_BYTES + 1024)   // 209920 B

__device__ __forceinline__ ushort_t f2bf(float f) {
  unsigned u = __float_as_uint(f);
  u += 0x7fffu + ((u >> 16) & 1u);            // round-to-nearest-even
  return (ushort_t)(u >> 16);
}
__device__ __forceinline__ float bf2f(ushort_t h) {
  return __uint_as_float(((unsigned)h) << 16);
}

// 16 contiguous bf16 (B operand run): two 16B LDS loads.
__device__ __forceinline__ v16bf ld16(const ushort_t* p) {
  v8bf a = *reinterpret_cast<const v8bf*>(p);
  v8bf b = *reinterpret_cast<const v8bf*>(p + 8);
  return __builtin_shufflevector(a, b, 0,1,2,3,4,5,6,7,8,9,10,11,12,13,14,15);
}
// Two 8-element runs 8 elements apart (16 contiguous; A operand, contiguous k order).
__device__ __forceinline__ v16bf ld2x8(const ushort_t* p) {
  v8bf a = *reinterpret_cast<const v8bf*>(p);
  v8bf b = *reinterpret_cast<const v8bf*>(p + 8);
  return __builtin_shufflevector(a, b, 0,1,2,3,4,5,6,7,8,9,10,11,12,13,14,15);
}

__device__ __forceinline__ v8f wmma_bf(v16bf a, v16bf b, v8f c) {
  return __builtin_amdgcn_wmma_f32_16x16x32_bf16(false, a, false, b, (short)0, c,
                                                 false, false);
}

__global__ __launch_bounds__(NTHREADS, 1)
void axial_attn_gfx1250(const float* __restrict__ Q,
                        const float* __restrict__ K,
                        const float* __restrict__ V,
                        float* __restrict__ O) {
  extern __shared__ unsigned char smem[];
  ushort_t* sKTh = (ushort_t*)(smem + KT_HI_OFF);
  ushort_t* sKTl = (ushort_t*)(smem + KT_LO_OFF);
  ushort_t* sVTh = (ushort_t*)(smem + VT_HI_OFF);
  ushort_t* sVTl = (ushort_t*)(smem + VT_LO_OFF);
  ushort_t* sPh  = (ushort_t*)(smem + QP_HI_OFF);   // Q hi, then P hi
  ushort_t* sPl  = (ushort_t*)(smem + QP_LO_OFF);   // Q lo, then P lo
  float*    sQsum = (float*)(smem + QSUM_OFF);
  float*    sKsum = (float*)(smem + KSUM_OFF);

  const int tid  = threadIdx.x;
  const int lane = tid & 31;
  const int w    = tid >> 5;                 // wave id 0..7 -> 16-row stripe
  const int b    = blockIdx.x >> 6;          // 64 L-tiles per batch
  const int tile = blockIdx.x & 63;

  const int m16 = lane & 15;
  const int hh  = lane >> 4;                 // lane half (ISA fragment layouts)
  const int aoff = hh ? 16 : 0;              // A-run offset within K-chunk
  const int boff = hh ? 16 : 0;              // B-run offset within K-chunk

  if (tid < S_N) { sQsum[tid] = 0.f; sKsum[tid] = 0.f; }
  __syncthreads();

  const float* Kb = K + (size_t)b * S_N * S_N;
  const float* Vb = V + (size_t)b * S_N * S_N;
  const float* Qb = Q + ((size_t)b * L_N + (size_t)tile * TILE_L) * S_N;

  // ---- Stage K^T (hi/lo bf16) + ksum[s] = sum_d keys[s][d] --------------
  for (int e = tid; e < S_N * S_N; e += NTHREADS) {
    int s = e >> 7, d = e & 127;             // coalesced global read
    float val = Kb[e];
    ushort_t hi = f2bf(val);
    ushort_t lo = f2bf(val - bf2f(hi));
    sKTh[d * SPAD + s] = hi;
    sKTl[d * SPAD + s] = lo;
    float ws = val;                          // all 32 lanes share the same s
    #pragma unroll
    for (int m = 1; m < 32; m <<= 1) ws += __shfl_xor(ws, m, 32);
    if (lane == 0) atomicAdd(&sKsum[s], ws);
  }
  // ---- Stage V^T (hi/lo bf16) -------------------------------------------
  for (int e = tid; e < S_N * S_N; e += NTHREADS) {
    int s = e >> 7, d = e & 127;
    float val = Vb[e];
    ushort_t hi = f2bf(val);
    ushort_t lo = f2bf(val - bf2f(hi));
    sVTh[d * SPAD + s] = hi;
    sVTl[d * SPAD + s] = lo;
  }
  // ---- Stage Q tile rows (hi/lo bf16) + qsum[r] -------------------------
  for (int e = tid; e < TILE_L * S_N; e += NTHREADS) {
    int r = e >> 7, c = e & 127;
    float val = Qb[e];
    ushort_t hi = f2bf(val);
    ushort_t lo = f2bf(val - bf2f(hi));
    sPh[r * SPAD + c] = hi;
    sPl[r * SPAD + c] = lo;
    float ws = val;
    #pragma unroll
    for (int m = 1; m < 32; m <<= 1) ws += __shfl_xor(ws, m, 32);
    if (lane == 0) atomicAdd(&sQsum[r], ws);
  }
  __syncthreads();

  // ---- Scores: S = Q·K via split-bf16 (Ah*Bh + Ah*Bl + Al*Bh) -----------
  v8f acc[8];
  #pragma unroll
  for (int cb = 0; cb < 8; ++cb) acc[cb] = (v8f){0.f,0.f,0.f,0.f,0.f,0.f,0.f,0.f};

  #pragma unroll
  for (int kk = 0; kk < 4; ++kk) {
    const ushort_t* pa = sPh + (w * 16 + m16) * SPAD + kk * 32 + aoff;
    const ushort_t* pl = sPl + (w * 16 + m16) * SPAD + kk * 32 + aoff;
    v16bf Ah = ld2x8(pa);
    v16bf Al = ld2x8(pl);
    #pragma unroll
    for (int cb = 0; cb < 8; ++cb) {
      const int brow = (cb * 16 + m16) * SPAD + kk * 32 + boff;
      v16bf Bh = ld16(sKTh + brow);
      v16bf Bl = ld16(sKTl + brow);
      acc[cb] = wmma_bf(Al, Bh, acc[cb]);
      acc[cb] = wmma_bf(Ah, Bl, acc[cb]);
      acc[cb] = wmma_bf(Ah, Bh, acc[cb]);
    }
  }

  // ---- Add rank-1 term s1 = qsum[l]*ksum[j]; row softmax (fp32) ---------
  float qs[8], mx[8], sm[8], rcp[8];
  #pragma unroll
  for (int v = 0; v < 8; ++v) qs[v] = sQsum[w * 16 + hh * 8 + v];
  #pragma unroll
  for (int cb = 0; cb < 8; ++cb) {
    float kc = sKsum[cb * 16 + m16];
    #pragma unroll
    for (int v = 0; v < 8; ++v) acc[cb][v] += qs[v] * kc;
  }
  #pragma unroll
  for (int v = 0; v < 8; ++v) {
    float m = acc[0][v];
    #pragma unroll
    for (int cb = 1; cb < 8; ++cb) m = fmaxf(m, acc[cb][v]);
    #pragma unroll
    for (int d = 1; d < 16; d <<= 1) m = fmaxf(m, __shfl_xor(m, d, 32));
    mx[v] = m;
    sm[v] = 0.f;
  }
  #pragma unroll
  for (int cb = 0; cb < 8; ++cb)
    #pragma unroll
    for (int v = 0; v < 8; ++v) {
      float p = __expf(acc[cb][v] - mx[v]);
      acc[cb][v] = p;                        // unnormalized P
      sm[v] += p;
    }
  #pragma unroll
  for (int v = 0; v < 8; ++v) {
    float s = sm[v];
    #pragma unroll
    for (int d = 1; d < 16; d <<= 1) s += __shfl_xor(s, d, 32);
    rcp[v] = 1.f / s;                        // fold normalization into output
  }

  // ---- Repack P (hi/lo bf16) into LDS, reusing the Q region ------------
  __syncthreads();                           // everyone done reading Q/K tiles
  #pragma unroll
  for (int cb = 0; cb < 8; ++cb) {
    int col = cb * 16 + m16;
    #pragma unroll
    for (int v = 0; v < 8; ++v) {
      int row = w * 16 + hh * 8 + v;         // C/D layout: M = v + 8*half
      float p = acc[cb][v];
      ushort_t hi = f2bf(p);
      ushort_t lo = f2bf(p - bf2f(hi));
      sPh[row * SPAD + col] = hi;
      sPl[row * SPAD + col] = lo;
    }
  }
  __syncthreads();

  // ---- out = P·V via split-bf16 -----------------------------------------
  v8f oacc[8];
  #pragma unroll
  for (int db = 0; db < 8; ++db) oacc[db] = (v8f){0.f,0.f,0.f,0.f,0.f,0.f,0.f,0.f};

  #pragma unroll
  for (int kk = 0; kk < 4; ++kk) {
    const ushort_t* pa = sPh + (w * 16 + m16) * SPAD + kk * 32 + aoff;
    const ushort_t* pl = sPl + (w * 16 + m16) * SPAD + kk * 32 + aoff;
    v16bf Ah = ld2x8(pa);
    v16bf Al = ld2x8(pl);
    #pragma unroll
    for (int db = 0; db < 8; ++db) {
      const int brow = (db * 16 + m16) * SPAD + kk * 32 + boff;
      v16bf Bh = ld16(sVTh + brow);
      v16bf Bl = ld16(sVTl + brow);
      oacc[db] = wmma_bf(Al, Bh, oacc[db]);
      oacc[db] = wmma_bf(Ah, Bl, oacc[db]);
      oacc[db] = wmma_bf(Ah, Bh, oacc[db]);
    }
  }

  // ---- Normalize and store (output is (B,1,L,D) == flat B*L*128) --------
  float* Ob = O + ((size_t)b * L_N + (size_t)tile * TILE_L) * S_N;
  #pragma unroll
  for (int db = 0; db < 8; ++db) {
    int col = db * 16 + m16;
    #pragma unroll
    for (int v = 0; v < 8; ++v) {
      int row = w * 16 + hh * 8 + v;
      Ob[(size_t)row * S_N + col] = oacc[db][v] * rcp[v];
    }
  }
}

extern "C" void kernel_launch(void* const* d_in, const int* in_sizes, int n_in,
                              void* d_out, int out_size, void* d_ws, size_t ws_size,
                              hipStream_t stream) {
  (void)in_sizes; (void)n_in; (void)out_size; (void)d_ws; (void)ws_size;
  const float* q = (const float*)d_in[0];
  const float* k = (const float*)d_in[1];
  const float* v = (const float*)d_in[2];
  float* out = (float*)d_out;

  dim3 grid(B_N * (L_N / TILE_L));           // 32 * 64 = 2048 workgroups
  dim3 block(NTHREADS);                      // 8 wave32 per WG
  axial_attn_gfx1250<<<grid, block, SMEM_BYTES, stream>>>(q, k, v, out);
}
